// SimTSC_2173253452203
// MI455X (gfx1250) — matrix operand order repaired and checked by reading, not compiled, not run on hardware
//
#include <hip/hip_runtime.h>
#include <hip/hip_bf16.h>
#include <math.h>

typedef __attribute__((ext_vector_type(16))) _Float16 v16h;
typedef __attribute__((ext_vector_type(8)))  _Float16 v8h;
typedef __attribute__((ext_vector_type(8)))  float    v8f;

#define NB   256
#define CH   512
#define LEN  512
#define ELEMS ((size_t)NB * CH * LEN)   // 67,108,864
#define EPS  1e-5f

// ---------------------------------------------------------------------------
// small utility kernels
// ---------------------------------------------------------------------------
__global__ __launch_bounds__(256) void zero_kernel(float* __restrict__ p, int count) {
    int i = blockIdx.x * 256 + threadIdx.x;
    if (i < count) p[i] = 0.f;
}

__global__ __launch_bounds__(256) void f32_to_f16_kernel(const float* __restrict__ s,
                                                         _Float16* __restrict__ d, int count) {
    int i = blockIdx.x * 256 + threadIdx.x;
    if (i < count) d[i] = (_Float16)s[i];
}

// global mean/var accumulators of x (131072 f32 elements) -> sx[0]=sum, sx[1]=sumsq
__global__ __launch_bounds__(256) void xstats_kernel(const float* __restrict__ x,
                                                     float* __restrict__ sx) {
    __shared__ float rs[256], rq[256];
    int i0 = blockIdx.x * 1024 + threadIdx.x;
    float s = 0.f, q = 0.f;
    #pragma unroll
    for (int t = 0; t < 4; ++t) { float v = x[i0 + t * 256]; s += v; q += v * v; }
    rs[threadIdx.x] = s; rq[threadIdx.x] = q;
    __syncthreads();
    for (int off = 128; off > 0; off >>= 1) {
        if (threadIdx.x < off) { rs[threadIdx.x] += rs[threadIdx.x + off];
                                 rq[threadIdx.x] += rq[threadIdx.x + off]; }
        __syncthreads();
    }
    if (threadIdx.x == 0) { atomicAdd(&sx[0], rs[0]); atomicAdd(&sx[1], rq[0]); }
}

// ---------------------------------------------------------------------------
// conv1: Cin=1, k=7, pad=3 (cheap, VALU)  -> f16 pre-BN output
// ---------------------------------------------------------------------------
__global__ __launch_bounds__(256) void conv1_kernel(const float* __restrict__ x,
                                                    const float* __restrict__ w,
                                                    const float* __restrict__ b,
                                                    _Float16* __restrict__ Y) {
    size_t idx = (size_t)blockIdx.x * 256 + threadIdx.x;
    int l = (int)(idx & 511);
    int c = (int)((idx >> 9) & 511);
    int n = (int)(idx >> 18);
    float acc = b[c];
    #pragma unroll
    for (int j = 0; j < 7; ++j) {
        int l2 = l + j - 3;
        if (l2 >= 0 && l2 < LEN) acc += w[c * 7 + j] * x[n * LEN + l2];
    }
    Y[idx] = (_Float16)acc;
}

// ---------------------------------------------------------------------------
// big conv: Cin=Cout=512, templated k, WMMA f16 -> f32 accumulate
// grid: (LEN/128, CH/64, NB), block 256 (8 waves = 4(co) x 2(l-groups of 64))
// each wave: 16co x 64l -> 4 accumulators; A fragment reused across 4 WMMAs
// ---------------------------------------------------------------------------
template <int K>
__global__ __launch_bounds__(256)
void conv_wmma_kernel(const _Float16* __restrict__ X, const _Float16* __restrict__ W,
                      const float* __restrict__ bias, _Float16* __restrict__ Y) {
    constexpr int PAD   = (K - 1) / 2;
    constexpr int LTILE = 128;
    constexpr int ROWS  = LTILE + K - 1;
    __shared__ __align__(32) _Float16 sW[K][64][32];   // [j][co][ci]
    __shared__ __align__(32) _Float16 sX[ROWS][32];    // [lcol][ci] (transposed)

    const int tid   = threadIdx.x;
    const int lane  = tid & 31;
    const int wv    = tid >> 5;
    const int coOff = (wv >> 1) * 16;     // 0,16,32,48
    const int lBase = (wv & 1) * 64;      // 0,64
    const int half  = lane >> 4;
    const int mr    = lane & 15;

    const int l0  = blockIdx.x * LTILE;
    const int co0 = blockIdx.y * 64;
    const int n   = blockIdx.z;

    v8f acc0 = {}, acc1 = {}, acc2 = {}, acc3 = {};
    for (int ci0 = 0; ci0 < CH; ci0 += 32) {
        // stage transposed input tile (covers all taps j at once)
        for (int idx = tid; idx < 32 * ROWS; idx += 256) {
            int ci = idx / ROWS;
            int r  = idx - ci * ROWS;
            int l  = l0 - PAD + r;
            _Float16 v = (_Float16)0.f;
            if (l >= 0 && l < LEN)
                v = X[((size_t)n * CH + (ci0 + ci)) * LEN + l];
            sX[r][ci] = v;
        }
        // stage weight tile
        for (int idx = tid; idx < K * 64 * 32; idx += 256) {
            int j   = idx >> 11;
            int rem = idx & 2047;
            int co  = rem >> 5;
            int ci  = rem & 31;
            sW[j][co][ci] = W[((size_t)(co0 + co) * CH + (ci0 + ci)) * K + j];
        }
        __syncthreads();
        #pragma unroll
        for (int j = 0; j < K; ++j) {
            // A fragment (16 co x 32 ci): lane row = mr, K = 16*half .. +15
            const _Float16* wrow = &sW[j][coOff + mr][0];
            v16h A;
            {
                v8h alo = *(const v8h*)(wrow + 16 * half);
                v8h ahi = *(const v8h*)(wrow + 16 * half + 8);
                #pragma unroll
                for (int i = 0; i < 8; ++i) { A[i] = alo[i]; A[i + 8] = ahi[i]; }
            }
            // B fragments (32 ci x 16 l): lane col = mr, K = 16*half .. +15
            const int rb = lBase + mr + j;
            v16h B0 = *(const v16h*)(&sX[rb +  0][0] + 16 * half);
            v16h B1 = *(const v16h*)(&sX[rb + 16][0] + 16 * half);
            v16h B2 = *(const v16h*)(&sX[rb + 32][0] + 16 * half);
            v16h B3 = *(const v16h*)(&sX[rb + 48][0] + 16 * half);
            acc0 = __builtin_amdgcn_wmma_f32_16x16x32_f16(false, A, false, B0, (short)0, acc0, false, false);
            acc1 = __builtin_amdgcn_wmma_f32_16x16x32_f16(false, A, false, B1, (short)0, acc1, false, false);
            acc2 = __builtin_amdgcn_wmma_f32_16x16x32_f16(false, A, false, B2, (short)0, acc2, false, false);
            acc3 = __builtin_amdgcn_wmma_f32_16x16x32_f16(false, A, false, B3, (short)0, acc3, false, false);
        }
        __syncthreads();
    }
    // C/D layout: N = mr (l), M = v + 8*half (co)
    const int co_b = co0 + coOff + 8 * half;
    v8f accs[4] = {acc0, acc1, acc2, acc3};
    #pragma unroll
    for (int t = 0; t < 4; ++t) {
        const int l_out = l0 + lBase + t * 16 + mr;
        #pragma unroll
        for (int v = 0; v < 8; ++v) {
            float val = accs[t][v] + bias[co_b + v];
            Y[((size_t)n * CH + (co_b + v)) * LEN + l_out] = (_Float16)val;
        }
    }
}

// ---------------------------------------------------------------------------
// per-channel sum / sumsq of an f16 activation tensor. grid (CH, NB), block 256
// ---------------------------------------------------------------------------
__global__ __launch_bounds__(256) void stats_kernel(const _Float16* __restrict__ T,
                                                    float* __restrict__ sbuf) {
    __shared__ float rs[256], rq[256];
    int c = blockIdx.x, n = blockIdx.y;
    size_t base = ((size_t)n * CH + c) * LEN;
    float a = (float)T[base + threadIdx.x];
    float b = (float)T[base + threadIdx.x + 256];
    rs[threadIdx.x] = a + b; rq[threadIdx.x] = a * a + b * b;
    __syncthreads();
    for (int off = 128; off > 0; off >>= 1) {
        if (threadIdx.x < off) { rs[threadIdx.x] += rs[threadIdx.x + off];
                                 rq[threadIdx.x] += rq[threadIdx.x + off]; }
        __syncthreads();
    }
    if (threadIdx.x == 0) { atomicAdd(&sbuf[c], rs[0]); atomicAdd(&sbuf[CH + c], rq[0]); }
}

// stats -> (scale, shift). <<<1,512>>>
__global__ void finalize_bn_kernel(const float* __restrict__ sbuf, const float* __restrict__ g,
                                   const float* __restrict__ beta, float* __restrict__ ss,
                                   float invCnt) {
    int c = threadIdx.x;
    float mean = sbuf[c] * invCnt;
    float var  = sbuf[CH + c] * invCnt - mean * mean;
    float sc   = g[c] * rsqrtf(var + EPS);
    ss[c]      = sc;
    ss[CH + c] = beta[c] - mean * sc;
}

// block1 shortcut BN folded analytically: bn(ws*x+bs) = sc*(x - mx), sc = gs*ws*rsqrt(ws^2*vx+eps)
__global__ void finalize_sc_kernel(const float* __restrict__ sx, const float* __restrict__ ws,
                                   const float* __restrict__ gs, const float* __restrict__ betas,
                                   float* __restrict__ ss) {
    int c = threadIdx.x;
    float mx = sx[0] * (1.f / 131072.f);
    float vx = sx[1] * (1.f / 131072.f) - mx * mx;
    float w  = ws[c];
    float sc = gs[c] * w * rsqrtf(w * w * vx + EPS);
    ss[c]      = sc;
    ss[CH + c] = betas[c] - sc * mx;
}

// elementwise BN + ReLU, in place, 8 f16 per thread (one b128 load/store)
__global__ __launch_bounds__(256) void bn_relu_kernel(_Float16* __restrict__ T,
                                                      const float* __restrict__ ss) {
    size_t i8 = ((size_t)blockIdx.x * 256 + threadIdx.x) * 8;
    int c = (int)((i8 >> 9) & 511);         // 8 | 512, all lanes share channel
    float sc = ss[c], sh = ss[CH + c];
    v8h v = *(v8h*)(T + i8);
    #pragma unroll
    for (int i = 0; i < 8; ++i) {
        float f = (float)v[i] * sc + sh;
        v[i] = (_Float16)(f > 0.f ? f : 0.f);
    }
    *(v8h*)(T + i8) = v;
}

// block1 tail: H = relu(bn(conv3) + bn(shortcut(x))); also accumulate H stats. grid (CH, NB)
__global__ __launch_bounds__(256)
void add_block1_kernel(const _Float16* __restrict__ P, const float* __restrict__ ssA,
                       const float* __restrict__ xin, const float* __restrict__ ssS,
                       _Float16* __restrict__ H, float* __restrict__ sH) {
    __shared__ float rs[256], rq[256];
    int c = blockIdx.x, n = blockIdx.y;
    size_t base = ((size_t)n * CH + c) * LEN;
    float sA = ssA[c], bA = ssA[CH + c];
    float sS = ssS[c], bS = ssS[CH + c];
    float s = 0.f, q = 0.f;
    #pragma unroll
    for (int t = 0; t < 2; ++t) {
        int l = threadIdx.x + t * 256;
        float v = (float)P[base + l] * sA + bA + xin[n * LEN + l] * sS + bS;
        v = v > 0.f ? v : 0.f;
        H[base + l] = (_Float16)v;
        s += v; q += v * v;
    }
    rs[threadIdx.x] = s; rq[threadIdx.x] = q;
    __syncthreads();
    for (int off = 128; off > 0; off >>= 1) {
        if (threadIdx.x < off) { rs[threadIdx.x] += rs[threadIdx.x + off];
                                 rq[threadIdx.x] += rq[threadIdx.x + off]; }
        __syncthreads();
    }
    if (threadIdx.x == 0) { atomicAdd(&sH[c], rs[0]); atomicAdd(&sH[CH + c], rq[0]); }
}

// block2 tail fused with global average pool: pooled[n,c] = mean_l relu(bn(P)+bn(H))
__global__ __launch_bounds__(256)
void add_block2_pool_kernel(const _Float16* __restrict__ P, const float* __restrict__ ssA,
                            const _Float16* __restrict__ H, const float* __restrict__ ssS,
                            float* __restrict__ pooled) {
    __shared__ float rs[256];
    int c = blockIdx.x, n = blockIdx.y;
    size_t base = ((size_t)n * CH + c) * LEN;
    float sA = ssA[c], bA = ssA[CH + c];
    float sS = ssS[c], bS = ssS[CH + c];
    float s = 0.f;
    #pragma unroll
    for (int t = 0; t < 2; ++t) {
        int l = threadIdx.x + t * 256;
        float v = (float)P[base + l] * sA + bA + (float)H[base + l] * sS + bS;
        s += (v > 0.f ? v : 0.f);
    }
    rs[threadIdx.x] = s;
    __syncthreads();
    for (int off = 128; off > 0; off >>= 1) {
        if (threadIdx.x < off) rs[threadIdx.x] += rs[threadIdx.x + off];
        __syncthreads();
    }
    if (threadIdx.x == 0) pooled[(size_t)n * CH + c] = rs[0] * (1.f / (float)LEN);
}

// head: logits = pooled @ w2^T + b2 ; log_softmax. one block per n
__global__ __launch_bounds__(256)
void head_kernel(const float* __restrict__ pooled, const float* __restrict__ w2,
                 const float* __restrict__ b2, float* __restrict__ out) {
    __shared__ float row[CH];
    __shared__ float lg[10];
    int n = blockIdx.x;
    row[threadIdx.x]       = pooled[(size_t)n * CH + threadIdx.x];
    row[threadIdx.x + 256] = pooled[(size_t)n * CH + threadIdx.x + 256];
    __syncthreads();
    if (threadIdx.x < 10) {
        float d = b2[threadIdx.x];
        for (int c = 0; c < CH; ++c) d += row[c] * w2[threadIdx.x * CH + c];
        lg[threadIdx.x] = d;
    }
    __syncthreads();
    if (threadIdx.x == 0) {
        float m = lg[0];
        for (int i = 1; i < 10; ++i) m = fmaxf(m, lg[i]);
        float s = 0.f;
        for (int i = 0; i < 10; ++i) s += expf(lg[i] - m);
        float lse = m + logf(s);
        for (int i = 0; i < 10; ++i) out[n * 10 + i] = lg[i] - lse;
    }
}

// ---------------------------------------------------------------------------
// launch
// ---------------------------------------------------------------------------
extern "C" void kernel_launch(void* const* d_in, const int* in_sizes, int n_in,
                              void* d_out, int out_size, void* d_ws, size_t ws_size,
                              hipStream_t stream) {
    (void)in_sizes; (void)n_in; (void)out_size; (void)ws_size;
    // input order: 0:x 1:adj 2:K 3:alpha
    // block1: 4:wx 5:bx 6:gx 7:betax 8:wy 9:by 10:gy 11:betay 12:wz 13:bz 14:gz 15:betaz
    //         16:gs 17:betas 18:ws 19:bs
    // block2: 20:wx 21:bx 22:gx 23:betax 24:wy 25:by 26:gy 27:betay 28:wz 29:bz 30:gz 31:betaz
    //         32:gs 33:betas
    // 34:w2 35:b2
    const float* x    = (const float*)d_in[0];
    const float* wx1  = (const float*)d_in[4];  const float* bx1 = (const float*)d_in[5];
    const float* gx1  = (const float*)d_in[6];  const float* betax1 = (const float*)d_in[7];
    const float* wy1  = (const float*)d_in[8];  const float* by1 = (const float*)d_in[9];
    const float* gy1  = (const float*)d_in[10]; const float* betay1 = (const float*)d_in[11];
    const float* wz1  = (const float*)d_in[12]; const float* bz1 = (const float*)d_in[13];
    const float* gz1  = (const float*)d_in[14]; const float* betaz1 = (const float*)d_in[15];
    const float* gs1  = (const float*)d_in[16]; const float* betas1 = (const float*)d_in[17];
    const float* ws1  = (const float*)d_in[18];
    const float* wx2  = (const float*)d_in[20]; const float* bx2 = (const float*)d_in[21];
    const float* gx2  = (const float*)d_in[22]; const float* betax2 = (const float*)d_in[23];
    const float* wy2  = (const float*)d_in[24]; const float* by2 = (const float*)d_in[25];
    const float* gy2  = (const float*)d_in[26]; const float* betay2 = (const float*)d_in[27];
    const float* wz2  = (const float*)d_in[28]; const float* bz2 = (const float*)d_in[29];
    const float* gz2  = (const float*)d_in[30]; const float* betaz2 = (const float*)d_in[31];
    const float* gs2  = (const float*)d_in[32]; const float* betas2 = (const float*)d_in[33];
    const float* w2   = (const float*)d_in[34]; const float* b2  = (const float*)d_in[35];
    float* out = (float*)d_out;

    // workspace carve-up
    const size_t szAct = ELEMS * sizeof(_Float16);            // 128 MB each
    char* base = (char*)d_ws;
    _Float16* P = (_Float16*)base;
    _Float16* Q = (_Float16*)(base + szAct);
    _Float16* H = (_Float16*)(base + 2 * szAct);
    char* wp = base + 3 * szAct;
    _Float16* wy1h = (_Float16*)wp; wp += (size_t)CH * CH * 5 * 2;
    _Float16* wz1h = (_Float16*)wp; wp += (size_t)CH * CH * 3 * 2;
    _Float16* wx2h = (_Float16*)wp; wp += (size_t)CH * CH * 7 * 2;
    _Float16* wy2h = (_Float16*)wp; wp += (size_t)CH * CH * 5 * 2;
    _Float16* wz2h = (_Float16*)wp; wp += (size_t)CH * CH * 3 * 2;
    float* stats = (float*)wp;
    float* sx  = stats;                                 // 2 (padded to 16)
    float* s0  = stats + 16 + 0 * 1024;                 // conv1 pre-BN stats
    float* s1  = stats + 16 + 1 * 1024;
    float* s2  = stats + 16 + 2 * 1024;
    float* s3  = stats + 16 + 3 * 1024;
    float* s4  = stats + 16 + 4 * 1024;
    float* s5  = stats + 16 + 5 * 1024;
    float* sH  = stats + 16 + 6 * 1024;
    float* ss0 = stats + 16 + 7 * 1024;                 // scale/shift pairs (no zeroing needed)
    float* ss1 = stats + 16 + 8 * 1024;
    float* ss2 = stats + 16 + 9 * 1024;
    float* ss3 = stats + 16 + 10 * 1024;
    float* ss4 = stats + 16 + 11 * 1024;
    float* ss5 = stats + 16 + 12 * 1024;
    float* ssc = stats + 16 + 13 * 1024;
    float* ssH = stats + 16 + 14 * 1024;
    float* pooled = stats + 16 + 15 * 1024;             // 131072 f32

    const float invCnt = 1.f / (float)(NB * LEN);       // 1/131072
    const int   zeroCnt = 16 + 7 * 1024;                // atomically-accumulated region
    const dim3  convGrid(LEN / 128, CH / 64, NB);
    const dim3  chanGrid(CH, NB);
    const int   ewGrid = (int)(ELEMS / 256);            // elementwise scalar kernels
    const int   bnGrid = (int)(ELEMS / (256 * 8));      // vectorized BN kernel

    zero_kernel<<<(zeroCnt + 255) / 256, 256, 0, stream>>>(stats, zeroCnt);
    f32_to_f16_kernel<<<(CH * CH * 5 + 255) / 256, 256, 0, stream>>>(wy1, wy1h, CH * CH * 5);
    f32_to_f16_kernel<<<(CH * CH * 3 + 255) / 256, 256, 0, stream>>>(wz1, wz1h, CH * CH * 3);
    f32_to_f16_kernel<<<(CH * CH * 7 + 255) / 256, 256, 0, stream>>>(wx2, wx2h, CH * CH * 7);
    f32_to_f16_kernel<<<(CH * CH * 5 + 255) / 256, 256, 0, stream>>>(wy2, wy2h, CH * CH * 5);
    f32_to_f16_kernel<<<(CH * CH * 3 + 255) / 256, 256, 0, stream>>>(wz2, wz2h, CH * CH * 3);
    xstats_kernel<<<128, 256, 0, stream>>>(x, sx);

    // ---- block 1 ----
    conv1_kernel<<<ewGrid, 256, 0, stream>>>(x, wx1, bx1, P);
    stats_kernel<<<chanGrid, 256, 0, stream>>>(P, s0);
    finalize_bn_kernel<<<1, 512, 0, stream>>>(s0, gx1, betax1, ss0, invCnt);
    bn_relu_kernel<<<bnGrid, 256, 0, stream>>>(P, ss0);

    conv_wmma_kernel<5><<<convGrid, 256, 0, stream>>>(P, wy1h, by1, Q);
    stats_kernel<<<chanGrid, 256, 0, stream>>>(Q, s1);
    finalize_bn_kernel<<<1, 512, 0, stream>>>(s1, gy1, betay1, ss1, invCnt);
    bn_relu_kernel<<<bnGrid, 256, 0, stream>>>(Q, ss1);

    conv_wmma_kernel<3><<<convGrid, 256, 0, stream>>>(Q, wz1h, bz1, P);
    stats_kernel<<<chanGrid, 256, 0, stream>>>(P, s2);
    finalize_bn_kernel<<<1, 512, 0, stream>>>(s2, gz1, betaz1, ss2, invCnt);
    finalize_sc_kernel<<<1, 512, 0, stream>>>(sx, ws1, gs1, betas1, ssc);
    add_block1_kernel<<<chanGrid, 256, 0, stream>>>(P, ss2, x, ssc, H, sH);

    // ---- block 2 ----
    conv_wmma_kernel<7><<<convGrid, 256, 0, stream>>>(H, wx2h, bx2, P);
    stats_kernel<<<chanGrid, 256, 0, stream>>>(P, s3);
    finalize_bn_kernel<<<1, 512, 0, stream>>>(s3, gx2, betax2, ss3, invCnt);
    bn_relu_kernel<<<bnGrid, 256, 0, stream>>>(P, ss3);

    conv_wmma_kernel<5><<<convGrid, 256, 0, stream>>>(P, wy2h, by2, Q);
    stats_kernel<<<chanGrid, 256, 0, stream>>>(Q, s4);
    finalize_bn_kernel<<<1, 512, 0, stream>>>(s4, gy2, betay2, ss4, invCnt);
    bn_relu_kernel<<<bnGrid, 256, 0, stream>>>(Q, ss4);

    conv_wmma_kernel<3><<<convGrid, 256, 0, stream>>>(Q, wz2h, bz2, P);
    stats_kernel<<<chanGrid, 256, 0, stream>>>(P, s5);
    finalize_bn_kernel<<<1, 512, 0, stream>>>(s5, gz2, betaz2, ss5, invCnt);
    finalize_bn_kernel<<<1, 512, 0, stream>>>(sH, gs2, betas2, ssH, invCnt);
    add_block2_pool_kernel<<<chanGrid, 256, 0, stream>>>(P, ss5, H, ssH, pooled);

    // ---- head ----
    head_kernel<<<NB, 256, 0, stream>>>(pooled, w2, b2, out);
}
